// SpatialAttentionGate_20650202759761
// MI455X (gfx1250) — matrix operand order repaired and checked by reading, not compiled, not run on hardware
//
#include <hip/hip_runtime.h>
#include <hip/hip_bf16.h>
#include <math.h>

// ---------------------------------------------------------------------------
// SAGAN spatial attention, fused for MI455X (gfx1250, wave32, WMMA).
//   B=8, C=256, H=W=64 -> N=4096, Cq=32
//   energy = (Wq x)^T (Wk x); A = softmax(energy, axis=-1); out = g * x A + x
// All heavy matmuls run on v_wmma_f32_16x16x32_f16. The N x N attention
// matrix is never materialized in HBM (recompute + online softmax stats).
// Workspace layout (d_ws):
//   [0,2MB)      Qh f16 [B][N][32]   (Q^T rows, contiguous in q)
//   [2,4MB)      Kh f16 [B][N][32]   (K^T rows, contiguous in q)
//   [4,20MB)     xh f16 [B][C][N]
//   [20MB,+128K) Mrow f32 [B][N]     (row max of energy)
//   [21MB,+128K) Rrow f32 [B][N]     (1 / row sum of exp)
// ---------------------------------------------------------------------------

#define BB 8
#define CC 256
#define NN 4096
#define CQ 32

typedef __attribute__((ext_vector_type(16))) _Float16 v16h;
typedef __attribute__((ext_vector_type(8)))  _Float16 v8h;
typedef __attribute__((ext_vector_type(8)))  float    v8f;

// Build a 16x(16-bit) WMMA fragment from two contiguous 16-byte chunks.
// Works for both global and LDS pointers (compiler infers addrspace).
__device__ __forceinline__ v16h load_frag(const _Float16* p0, const _Float16* p1) {
  v8h lo = *(const v8h*)p0;
  v8h hi = *(const v8h*)p1;
  v16h r;
#pragma unroll
  for (int i = 0; i < 8; ++i) { r[i] = lo[i]; r[i + 8] = hi[i]; }
  return r;
}

__device__ __forceinline__ v8f wmma_f16(v16h a, v16h b, v8f c) {
  return __builtin_amdgcn_wmma_f32_16x16x32_f16(false, a, false, b, (short)0, c,
                                                false, false);
}

// ---------------------------------------------------------------------------
// Kernel 1: Q/K 1x1-conv projections (f32 FMA, tiny) + f16 casts.
// grid = B * (N/64), block = 256. Thread t: q = t>>3, handles 8 n positions.
// ---------------------------------------------------------------------------
__global__ void __launch_bounds__(256)
k_proj_cast(const float* __restrict__ x, const float* __restrict__ wq,
            const float* __restrict__ bq, const float* __restrict__ wk,
            const float* __restrict__ bk, _Float16* __restrict__ Qh,
            _Float16* __restrict__ Kh, _Float16* __restrict__ xh) {
  __shared__ float xs[64][65];
  const int b  = blockIdx.x >> 6;
  const int n0 = (blockIdx.x & 63) << 6;
  const int t  = threadIdx.x;
  const int q = t >> 3, nsub = t & 7;

  float aq[8], ak[8];
#pragma unroll
  for (int i = 0; i < 8; ++i) { aq[i] = 0.f; ak[i] = 0.f; }

  for (int c0 = 0; c0 < CC; c0 += 64) {
    __syncthreads();
    // stage x[b, c0:c0+64, n0:n0+64] into LDS; also emit f16 copy of x
#pragma unroll
    for (int idx = t; idx < 64 * 64; idx += 256) {
      const int cc = idx >> 6, nn = idx & 63;
      const size_t gi = (size_t)(b * CC + c0 + cc) * NN + n0 + nn;
      const float v = x[gi];
      xs[cc][nn] = v;
      xh[gi] = (_Float16)v;
    }
    __syncthreads();
#pragma unroll 4
    for (int cc = 0; cc < 64; ++cc) {
      const float wqv = wq[q * CC + c0 + cc];
      const float wkv = wk[q * CC + c0 + cc];
#pragma unroll
      for (int i = 0; i < 8; ++i) {
        const float xv = xs[cc][nsub + 8 * i];
        aq[i] = fmaf(wqv, xv, aq[i]);
        ak[i] = fmaf(wkv, xv, ak[i]);
      }
    }
  }
  const float bqv = bq[q], bkv = bk[q];
#pragma unroll
  for (int i = 0; i < 8; ++i) {
    const int n = n0 + nsub + 8 * i;
    const size_t o = (size_t)(b * NN + n) * CQ + q;
    Qh[o] = (_Float16)(aq[i] + bqv);
    Kh[o] = (_Float16)(ak[i] + bkv);
  }
}

// ---------------------------------------------------------------------------
// Kernel 2: per-row softmax stats (m_j, 1/S_j) over the full energy row.
// One wave per 16-row j-tile; 256 WMMAs sweep all k. Online max/sum in the
// C/D layout, then xor-shuffle reduce across the 16-lane column group.
// grid = B * 32, block = 256 (8 waves, 8 j-tiles per block).
// ---------------------------------------------------------------------------
__global__ void __launch_bounds__(256)
k_rowstats(const _Float16* __restrict__ Qh, const _Float16* __restrict__ Kh,
           float* __restrict__ Mrow, float* __restrict__ Rrow) {
  const int lane = threadIdx.x & 31, wave = threadIdx.x >> 5;
  const int b = blockIdx.x >> 5;
  const int jt = ((blockIdx.x & 31) << 3) + wave;  // 0..255
  const int j0 = jt << 4;
  const int hg = lane >> 4, lm = lane & 15;

  // A fragment: Q^T rows j0+lm, all 32 q (loop-invariant)
  const _Float16* qb = Qh + (size_t)(b * NN + j0 + lm) * CQ + 8 * hg;
  const v16h aq = load_frag(qb, qb + 16);
  const v8f zero = {};

  float m[8], s[8];
#pragma unroll
  for (int r = 0; r < 8; ++r) { m[r] = -INFINITY; s[r] = 0.f; }

  for (int k0 = 0; k0 < NN; k0 += 16) {
    const _Float16* kb = Kh + (size_t)(b * NN + k0 + lm) * CQ + 8 * hg;
    const v16h bk = load_frag(kb, kb + 16);
    const v8f e = wmma_f16(aq, bk, zero);
#pragma unroll
    for (int r = 0; r < 8; ++r) {
      const float ev = e[r];
      const float mo = m[r];
      const float mn = fmaxf(mo, ev);
      s[r] = s[r] * __expf(mo - mn) + __expf(ev - mn);
      m[r] = mn;
    }
  }
  // combine (m,s) across the 16 lanes holding the same rows (xor < 16)
#pragma unroll
  for (int d = 1; d < 16; d <<= 1) {
#pragma unroll
    for (int r = 0; r < 8; ++r) {
      const float mo = __shfl_xor(m[r], d, 32);
      const float so = __shfl_xor(s[r], d, 32);
      const float mn = fmaxf(m[r], mo);
      s[r] = s[r] * __expf(m[r] - mn) + so * __expf(mo - mn);
      m[r] = mn;
    }
  }
  if (lm == 0) {
#pragma unroll
    for (int r = 0; r < 8; ++r) {
      const int j = b * NN + j0 + 8 * hg + r;
      Mrow[j] = m[r];
      Rrow[j] = 1.0f / s[r];
    }
  }
}

// ---------------------------------------------------------------------------
// Kernel 3: fused  out[:, k-tile] = gamma * (x @ softmax(Q^T K)) + x.
// grid = B * (N/64), block = 256 (8 waves). Per 32-wide j chunk:
//   wave w computes energy tile (jsub = w&1, ksub = w>>1) -> P (f16) in a
//   double-buffered LDS tile (ONE barrier per iteration), then all waves run
//   8 accumulate WMMAs (2 c-tiles x 4 k-subtiles). Softmax stats are staged
//   in LDS once per block (32 KB, L2 -> LDS) so the hot loop only does
//   contiguous 16B ds loads for them.
// ---------------------------------------------------------------------------
__global__ void __launch_bounds__(256)
k_attn_out(const _Float16* __restrict__ Qh, const _Float16* __restrict__ Kh,
           const _Float16* __restrict__ xh, const float* __restrict__ Mrow,
           const float* __restrict__ Rrow, const float* __restrict__ x,
           const float* __restrict__ gamma, float* __restrict__ out) {
  __shared__ _Float16 Pt[2][64 * 40];  // [k_local][j_local], row = 80B
  __shared__ float Ms[NN];             // 16 KB: row max, this batch
  __shared__ float Rs[NN];             // 16 KB: 1/rowsum, this batch

  const int lane = threadIdx.x & 31, wave = threadIdx.x >> 5;
  const int b = blockIdx.x >> 6;
  const int k0blk = (blockIdx.x & 63) << 6;  // 64-wide k tile
  const int hg = lane >> 4, lm = lane & 15;
  const int jsub = wave & 1, ksw = wave >> 1;

  // Stage the softmax stats for this batch into LDS (coalesced float4).
  {
    const float4* ms = (const float4*)(Mrow + b * NN);
    const float4* rs = (const float4*)(Rrow + b * NN);
    float4* ml = (float4*)Ms;
    float4* rl = (float4*)Rs;
#pragma unroll
    for (int i = threadIdx.x; i < NN / 4; i += 256) {
      ml[i] = ms[i];
      rl[i] = rs[i];
    }
  }

  // Loop-invariant B fragment for the energy WMMA: K columns of this wave's
  // k-subtile. Kh[b][k][q], rows q = 8*hg + {0..7} and 16 + 8*hg + {0..7}.
  const _Float16* kb =
      Kh + (size_t)(b * NN + k0blk + 16 * ksw + lm) * CQ + 8 * hg;
  const v16h bkf = load_frag(kb, kb + 16);

  const v8f zero = {};
  v8f acc[8];
#pragma unroll
  for (int i = 0; i < 8; ++i) acc[i] = zero;

  __syncthreads();  // stats staged

  int buf = 0;
  for (int j0 = 0; j0 < NN; j0 += 32, buf ^= 1) {
    // --- energy tile for (jsub, ksw) ---
    const int jrow = j0 + 16 * jsub + lm;
    const _Float16* qb = Qh + (size_t)(b * NN + jrow) * CQ + 8 * hg;
    const v16h aq = load_frag(qb, qb + 16);
    const v8f e = wmma_f16(aq, bkf, zero);

    if (j0 + 32 < NN)  // prefetch next chunk's Q rows
      __builtin_prefetch((const void*)(qb + 32 * CQ), 0, 0);

    // --- P = exp(e - m) * invS, packed f16, store transposed tile to LDS ---
    const int jbase = j0 + 16 * jsub + 8 * hg;  // multiple of 8
    const float4 m0 = *(const float4*)&Ms[jbase];
    const float4 m1 = *(const float4*)&Ms[jbase + 4];
    const float4 r0 = *(const float4*)&Rs[jbase];
    const float4 r1 = *(const float4*)&Rs[jbase + 4];
    v8h p;
    p[0] = (_Float16)(__expf(e[0] - m0.x) * r0.x);
    p[1] = (_Float16)(__expf(e[1] - m0.y) * r0.y);
    p[2] = (_Float16)(__expf(e[2] - m0.z) * r0.z);
    p[3] = (_Float16)(__expf(e[3] - m0.w) * r0.w);
    p[4] = (_Float16)(__expf(e[4] - m1.x) * r1.x);
    p[5] = (_Float16)(__expf(e[5] - m1.y) * r1.y);
    p[6] = (_Float16)(__expf(e[6] - m1.z) * r1.z);
    p[7] = (_Float16)(__expf(e[7] - m1.w) * r1.w);

    const int kl = 16 * ksw + lm;
    *(v8h*)&Pt[buf][kl * 40 + 16 * jsub + 8 * hg] = p;
    __syncthreads();  // sole barrier: P(buf) visible; prior reads of buf done

    // --- accumulate out += xh[:, j-chunk] @ P ---
    v16h bp[4];
#pragma unroll
    for (int ks = 0; ks < 4; ++ks) {
      const _Float16* pb = &Pt[buf][(16 * ks + lm) * 40 + 8 * hg];
      bp[ks] = load_frag(pb, pb + 16);
    }
#pragma unroll
    for (int ct = 0; ct < 2; ++ct) {
      const int ctile = wave + 8 * ct;
      const _Float16* xb =
          xh + (size_t)(b * CC + 16 * ctile + lm) * NN + j0 + 8 * hg;
      const v16h ax = load_frag(xb, xb + 16);
      if (j0 + 32 < NN)
        __builtin_prefetch((const void*)(xb + 32), 0, 0);
#pragma unroll
      for (int ks = 0; ks < 4; ++ks)
        acc[ct * 4 + ks] = wmma_f16(ax, bp[ks], acc[ct * 4 + ks]);
    }
  }

  // --- epilogue: out = gamma * acc + x, straight from C/D layout ---
  const float g = gamma[0];
#pragma unroll
  for (int ct = 0; ct < 2; ++ct) {
    const int ctile = wave + 8 * ct;
#pragma unroll
    for (int ks = 0; ks < 4; ++ks) {
      const v8f a = acc[ct * 4 + ks];
#pragma unroll
      for (int r = 0; r < 8; ++r) {
        const int c = 16 * ctile + 8 * hg + r;
        const int k = k0blk + 16 * ks + lm;
        const size_t idx = (size_t)(b * CC + c) * NN + k;
        out[idx] = fmaf(g, a[r], x[idx]);
      }
    }
  }
}

// ---------------------------------------------------------------------------
extern "C" void kernel_launch(void* const* d_in, const int* in_sizes, int n_in,
                              void* d_out, int out_size, void* d_ws,
                              size_t ws_size, hipStream_t stream) {
  (void)in_sizes; (void)n_in; (void)out_size; (void)ws_size;
  const float* x     = (const float*)d_in[0];
  const float* wq    = (const float*)d_in[1];
  const float* bq    = (const float*)d_in[2];
  const float* wk    = (const float*)d_in[3];
  const float* bk    = (const float*)d_in[4];
  const float* gamma = (const float*)d_in[5];
  float* out = (float*)d_out;

  char* ws = (char*)d_ws;
  _Float16* Qh = (_Float16*)(ws);
  _Float16* Kh = (_Float16*)(ws + (size_t)2 * 1024 * 1024);
  _Float16* xh = (_Float16*)(ws + (size_t)4 * 1024 * 1024);
  float* Mrow  = (float*)(ws + (size_t)20 * 1024 * 1024);
  float* Rrow  = (float*)(ws + (size_t)21 * 1024 * 1024);

  k_proj_cast<<<BB * (NN / 64), 256, 0, stream>>>(x, wq, bq, wk, bk, Qh, Kh, xh);
  k_rowstats<<<BB * 32, 256, 0, stream>>>(Qh, Kh, Mrow, Rrow);
  k_attn_out<<<BB * (NN / 64), 256, 0, stream>>>(Qh, Kh, xh, Mrow, Rrow, x,
                                                 gamma, out);
}
